// DeepRadAEVComputer_18124761989630
// MI455X (gfx1250) — matrix operand order repaired and checked by reading, not compiled, not run on hardware
//
#include <hip/hip_runtime.h>
#include <hip/hip_bf16.h>
#include <stdint.h>

typedef __attribute__((ext_vector_type(16))) _Float16 v16h;
typedef __attribute__((ext_vector_type(8)))  _Float16 v8h;
typedef __attribute__((ext_vector_type(8)))  float    v8f;

#define XS      72          // LDS activation/weight row stride (halves): 144B, 16B-aligned, bank-spread
#define LREG    (64 * XS)   // halves per 64-wide weight region
#define W6BASE  (6 * LREG)  // offset of layer-6 weights (128 rows)
#define WIMG_BYTES ((W6BASE + 128 * XS) * 2)   // 73,728 B f16 weight image
#define BIAS_BYTES (448 * 4)                   // 1,792 B f32 bias image
#define NPAIR   256
#define CUTOFF_F 5.2f
#define EPS_F    1e-7f
#define PI_OVER_CUT 0.604152433f   // pi / 5.2
#define TWO_LOG2E 2.885390082f     // 2 * log2(e)

// Intra-wave ordering only: CDNA5 LDS ops are in-order per wave (DScnt),
// so layer boundaries need just a compiler fence, not a workgroup barrier.
#define WAVE_FENCE() asm volatile("" ::: "memory")

// Branch-free fast tanh: 1 - 2/(exp2(x*2log2e)+1). Saturates at +-1, no NaN path.
__device__ __forceinline__ float ftanh(float x) {
    float e = __builtin_amdgcn_exp2f(x * TWO_LOG2E);
    float r = __builtin_amdgcn_rcpf(e + 1.0f);
    return fmaf(-2.0f, r, 1.0f);
}

__device__ __forceinline__ v8f wmma16(v16h a, v16h b, v8f c) {
    // D = A(16x32 f16) * B(32x16 f16) + C(16x16 f32)
    return __builtin_amdgcn_wmma_f32_16x16x32_f16(
        false, a, false, b, (short)0, c, false, false);
}

// A-matrix (16x32 f16) per ISA 7.12.2: lanes 0-15 row M=lane, K {0..7,16..23};
// lanes 16-31 row M=lane-16, K {8..15,24..31}. Activations row-major in LDS.
__device__ __forceinline__ v16h ld_a(const _Float16* x, int mbase, int kb, int lane) {
    int row = mbase + (lane & 15);
    const _Float16* p = x + row * XS + kb + ((lane & 16) ? 8 : 0);
    v8h lo = *(const v8h*)p;          // K = kb+kh .. +7
    v8h hi = *(const v8h*)(p + 16);   // K = kb+16+kh .. +7
    return __builtin_shufflevector(lo, hi, 0,1,2,3,4,5,6,7,8,9,10,11,12,13,14,15);
}

// B-matrix (32x16 f16): lane holds column N=lane&15; lanes 0-15 K {0..7,16..23},
// lanes 16-31 K {8..15,24..31}. Weights stored transposed: wl[n*XS + k].
__device__ __forceinline__ v16h ld_b(const _Float16* wl, int nb, int kb, int lane) {
    int n = nb + (lane & 15);
    const _Float16* p = wl + n * XS + kb + ((lane & 16) ? 8 : 0);
    v8h lo = *(const v8h*)p;
    v8h hi = *(const v8h*)(p + 16);
    return __builtin_shufflevector(lo, hi, 0,1,2,3,4,5,6,7,8,9,10,11,12,13,14,15);
}

// Write a 16x16 f32 C-tile back to LDS as f16 activations (row-major).
__device__ __forceinline__ void st_c(_Float16* x, int mbase, int nt, int lane, v8f v) {
    int rbase = mbase + ((lane & 16) ? 8 : 0);
    int col = nt * 16 + (lane & 15);
#pragma unroll
    for (int r = 0; r < 8; ++r)
        x[(rbase + r) * XS + col] = (_Float16)v[r];
}

// ---------------------------------------------------------------------------
// Prep kernel: convert all weights ONCE into d_ws as the exact f16 LDS image
// (transposed, XS-stride, layer-0 K-padded with bias row), biases appended f32.
// ---------------------------------------------------------------------------
__global__ __launch_bounds__(256)
void DeepRad_prep_kernel(
    const float* __restrict__ W0, const float* __restrict__ b0,
    const float* __restrict__ W1, const float* __restrict__ b1,
    const float* __restrict__ W2, const float* __restrict__ b2,
    const float* __restrict__ W3, const float* __restrict__ b3,
    const float* __restrict__ W4, const float* __restrict__ b4,
    const float* __restrict__ W5, const float* __restrict__ b5,
    const float* __restrict__ W6, const float* __restrict__ b6,
    _Float16* __restrict__ wsW, float* __restrict__ wsB)
{
    const int gid = blockIdx.x * 256 + threadIdx.x;
    const int gs  = gridDim.x * 256;

    // Layer 0: K padded to 32; k<3 = W0 rows, k==3 = bias (input col 3 == 1.0)
    for (int idx = gid; idx < 64 * 32; idx += gs) {
        int n = idx >> 5, k = idx & 31;
        float v = 0.0f;
        if (k < 3)       v = W0[k * 64 + n];
        else if (k == 3) v = b0[n];
        wsW[n * XS + k] = (_Float16)v;
    }
    const float* Wm[5] = {W1, W2, W3, W4, W5};
    const float* bm[5] = {b1, b2, b3, b4, b5};
    for (int l = 0; l < 5; ++l) {
        _Float16* wl = wsW + (l + 1) * LREG;
        for (int idx = gid; idx < 64 * 64; idx += gs) {
            int k = idx >> 6, n = idx & 63;
            wl[n * XS + k] = (_Float16)Wm[l][k * 64 + n];
        }
    }
    for (int idx = gid; idx < 64 * 128; idx += gs) {
        int k = idx >> 7, n = idx & 127;
        wsW[W6BASE + n * XS + k] = (_Float16)W6[k * 128 + n];
    }
    for (int idx = gid; idx < 448; idx += gs)
        wsB[idx] = (idx < 320) ? bm[idx >> 6][idx & 63] : b6[idx - 320];
}

// ---------------------------------------------------------------------------
// Main kernel: one block per (b,i); 8 wave32s; WMMA f16 MLP over j=0..255.
// ---------------------------------------------------------------------------
__global__ __launch_bounds__(256)
void DeepRadAEVComputer_18124761989630_kernel(
    const float* __restrict__ D, const int* __restrict__ Z,
    const _Float16* __restrict__ wsW, float* __restrict__ out)
{
    __shared__ __attribute__((aligned(16))) _Float16 sX[NPAIR * XS];          // 36,864 B
    __shared__ __attribute__((aligned(16))) _Float16 sW[W6BASE + 128 * XS];   // 73,728 B
    __shared__ __attribute__((aligned(16))) float sBias[448];
    __shared__ float sWrow[NPAIR];
    __shared__ float sGR[128];
    __shared__ float sRed[64];

    const int tid = threadIdx.x;
    const int bi  = blockIdx.x;          // b*256 + i
    const int bb  = bi >> 8;
    const int ii  = bi & 255;

    // ---- stage prepacked weight image via async global->LDS DMA -----------
    {
        const uint64_t gbase = (uint64_t)(uintptr_t)wsW;
        const uint32_t ldsW  = (uint32_t)(uintptr_t)sW;   // LDS aperture: addr[31:0] = LDS offset
        const uint32_t ldsB  = (uint32_t)(uintptr_t)sBias;
#pragma unroll 1
        for (int off = tid * 16; off < WIMG_BYTES; off += 256 * 16) {
            uint32_t la = ldsW + (uint32_t)off;
            uint64_t ga = gbase + (uint64_t)off;
            asm volatile("global_load_async_to_lds_b128 %0, %1, off"
                         :: "v"(la), "v"(ga) : "memory");
        }
        if (tid < BIAS_BYTES / 16) {
            uint32_t la = ldsB + (uint32_t)(tid * 16);
            uint64_t ga = gbase + (uint64_t)WIMG_BYTES + (uint64_t)(tid * 16);
            asm volatile("global_load_async_to_lds_b128 %0, %1, off"
                         :: "v"(la), "v"(ga) : "memory");
        }
    }
    if (tid < 128) sGR[tid] = 0.0f;

    // ---- build MLP input row j = tid --------------------------------------
    {
        const int j = tid;
        float d  = D[bi * NPAIR + j];
        float zi = (float)Z[bb * NPAIR + ii];
        float zj = (float)Z[bb * NPAIR + j];
        float s  = zi + zj, pr = zi * zj;
        float inv = 1.0f / (sqrtf(s * s + pr * pr) + EPS_F);
        _Float16* xr = sX + j * XS;
        xr[0] = (_Float16)d;
        xr[1] = (_Float16)(s * inv);
        xr[2] = (_Float16)(pr * inv);
        xr[3] = (_Float16)1.0f;          // carries layer-0 bias
#pragma unroll
        for (int k = 4; k < 32; ++k) xr[k] = (_Float16)0.0f;
        float fc = 0.5f * __cosf(PI_OVER_CUT * d) + 0.5f;
        sWrow[j] = (d < CUTOFF_F && d != 0.0f) ? fc : 0.0f;
    }
    asm volatile("s_wait_asynccnt 0x0" ::: "memory");
    __syncthreads();   // cross-wave: weight image staged by all waves

    const int wave = tid >> 5;
    const int lane = tid & 31;
    const int mb   = wave * 32;          // this wave's first activation row

    v8f res[2][4];
    v8f acc[2][4];

    // ---- layer 0: (256x32) @ (32x64), bias folded -------------------------
#pragma unroll
    for (int mt = 0; mt < 2; ++mt)
#pragma unroll
        for (int nt = 0; nt < 4; ++nt)
#pragma unroll
            for (int e = 0; e < 8; ++e) acc[mt][nt][e] = 0.0f;
    {
        v16h a0 = ld_a(sX, mb,      0, lane);
        v16h a1 = ld_a(sX, mb + 16, 0, lane);
#pragma unroll
        for (int nt = 0; nt < 4; ++nt) {
            v16h bt = ld_b(sW, nt * 16, 0, lane);
            acc[0][nt] = wmma16(a0, bt, acc[0][nt]);
            acc[1][nt] = wmma16(a1, bt, acc[1][nt]);
        }
    }
#pragma unroll
    for (int mt = 0; mt < 2; ++mt)
#pragma unroll
        for (int nt = 0; nt < 4; ++nt) {
            v8f o;
#pragma unroll
            for (int e = 0; e < 8; ++e) {
                float t = ftanh(acc[mt][nt][e]);
                res[mt][nt][e] = t;      // h0 starts residual stream
                o[e] = t;
            }
            st_c(sX, mb + mt * 16, nt, lane, o);
        }
    WAVE_FENCE();   // wave-private rows; HW keeps per-wave LDS ops in order

    // ---- layers 1..5: (256x64) @ (64x64), fully unrolled ------------------
#pragma unroll
    for (int l = 1; l <= 5; ++l) {
        const _Float16* wl = sW + l * LREG;
#pragma unroll
        for (int mt = 0; mt < 2; ++mt)
#pragma unroll
            for (int nt = 0; nt < 4; ++nt)
#pragma unroll
                for (int e = 0; e < 8; ++e) acc[mt][nt][e] = 0.0f;
#pragma unroll
        for (int kt = 0; kt < 2; ++kt) {
            int kb = kt * 32;
            v16h a0 = ld_a(sX, mb,      kb, lane);
            v16h a1 = ld_a(sX, mb + 16, kb, lane);
#pragma unroll
            for (int nt = 0; nt < 4; ++nt) {
                v16h bt = ld_b(wl, nt * 16, kb, lane);
                acc[0][nt] = wmma16(a0, bt, acc[0][nt]);
                acc[1][nt] = wmma16(a1, bt, acc[1][nt]);
            }
        }
        const int boff = (l - 1) * 64;
        const int addres = (l & 1);      // layers 1,3,5 add+update residual
#pragma unroll
        for (int mt = 0; mt < 2; ++mt)
#pragma unroll
            for (int nt = 0; nt < 4; ++nt) {
                float bv = sBias[boff + nt * 16 + (lane & 15)];
                v8f o;
#pragma unroll
                for (int e = 0; e < 8; ++e) {
                    float t = ftanh(acc[mt][nt][e] + bv);
                    if (addres) { t += res[mt][nt][e]; res[mt][nt][e] = t; }
                    o[e] = t;
                }
                st_c(sX, mb + mt * 16, nt, lane, o);
            }
        WAVE_FENCE();
    }

    // ---- layer 6: (256x64) @ (64x128) + cutoff-weighted j-reduction -------
    float wv[2][8];
#pragma unroll
    for (int mt = 0; mt < 2; ++mt)
#pragma unroll
        for (int e = 0; e < 8; ++e)
            wv[mt][e] = sWrow[mb + mt * 16 + ((lane & 16) ? 8 : 0) + e];

#pragma unroll
    for (int p = 0; p < 2; ++p) {        // two passes of 4 N-tiles each
#pragma unroll
        for (int mt = 0; mt < 2; ++mt)
#pragma unroll
            for (int nt = 0; nt < 4; ++nt)
#pragma unroll
                for (int e = 0; e < 8; ++e) acc[mt][nt][e] = 0.0f;
#pragma unroll
        for (int kt = 0; kt < 2; ++kt) {
            int kb = kt * 32;
            v16h a0 = ld_a(sX, mb,      kb, lane);
            v16h a1 = ld_a(sX, mb + 16, kb, lane);
#pragma unroll
            for (int nt = 0; nt < 4; ++nt) {
                int ntg = p * 4 + nt;
                v16h bt = ld_b(sW + W6BASE, ntg * 16, kb, lane);
                acc[0][nt] = wmma16(a0, bt, acc[0][nt]);
                acc[1][nt] = wmma16(a1, bt, acc[1][nt]);
            }
        }
#pragma unroll
        for (int nt = 0; nt < 4; ++nt) {
            int ntg = p * 4 + nt;
            float bv = sBias[320 + ntg * 16 + (lane & 15)];
            float partial = 0.0f;
#pragma unroll
            for (int mt = 0; mt < 2; ++mt)
#pragma unroll
                for (int e = 0; e < 8; ++e)
                    partial += ftanh(acc[mt][nt][e] + bv) * wv[mt][e];
            atomicAdd(&sGR[ntg * 16 + (lane & 15)], partial);
        }
    }
    __syncthreads();   // cross-wave: all atomic partials landed

    // ---- normalize GR row and store ---------------------------------------
    if (tid < 64) {
        float a = sGR[tid], c = sGR[tid + 64];
        sRed[tid] = a * a + c * c;
    }
    __syncthreads();
    for (int off = 32; off > 0; off >>= 1) {
        if (tid < off) sRed[tid] += sRed[tid + off];
        __syncthreads();
    }
    if (tid < 128)
        out[bi * 128 + tid] = sGR[tid] / (sqrtf(sRed[0]) + EPS_F);
}

extern "C" void kernel_launch(void* const* d_in, const int* in_sizes, int n_in,
                              void* d_out, int out_size, void* d_ws, size_t ws_size,
                              hipStream_t stream) {
    (void)in_sizes; (void)n_in; (void)out_size; (void)ws_size;
    const float* D  = (const float*)d_in[0];
    const int*   Z  = (const int*)d_in[1];
    const float* W0 = (const float*)d_in[2];  const float* b0 = (const float*)d_in[3];
    const float* W1 = (const float*)d_in[4];  const float* b1 = (const float*)d_in[5];
    const float* W2 = (const float*)d_in[6];  const float* b2 = (const float*)d_in[7];
    const float* W3 = (const float*)d_in[8];  const float* b3 = (const float*)d_in[9];
    const float* W4 = (const float*)d_in[10]; const float* b4 = (const float*)d_in[11];
    const float* W5 = (const float*)d_in[12]; const float* b5 = (const float*)d_in[13];
    const float* W6 = (const float*)d_in[14]; const float* b6 = (const float*)d_in[15];
    float* out = (float*)d_out;

    _Float16* wsW = (_Float16*)d_ws;                       // f16 weight image (XS-stride, transposed)
    float*    wsB = (float*)((char*)d_ws + WIMG_BYTES);    // f32 bias image

    DeepRad_prep_kernel<<<16, 256, 0, stream>>>(
        W0, b0, W1, b1, W2, b2, W3, b3, W4, b4, W5, b5, W6, b6, wsW, wsB);

    // one block per (b, i): 8 * 256 = 2048 blocks, 256 threads = 8 wave32s
    DeepRadAEVComputer_18124761989630_kernel<<<2048, 256, 0, stream>>>(D, Z, wsW, out);
}
